// Backbone_11776800326350
// MI455X (gfx1250) — compile-verified, matrix operand order checked by reading, not run on hardware
//
#include <hip/hip_runtime.h>
#include <hip/hip_bf16.h>

typedef __attribute__((ext_vector_type(16))) __bf16 v16bf;
typedef __attribute__((ext_vector_type(8)))  float  v8f;

#define DIM 128
#define NEG_SLOPE 0.01f

// ---------------------------------------------------------------- utilities
__global__ __launch_bounds__(256) void k_fill(float* __restrict__ p, size_t n, float v) {
    size_t i = (size_t)blockIdx.x * blockDim.x + threadIdx.x;
    if (i < n) p[i] = v;
}

__global__ __launch_bounds__(256) void k_deg(const long long* __restrict__ ei,
                                             float* __restrict__ deg, int E) {
    int e = blockIdx.x * blockDim.x + threadIdx.x;
    if (e < E) atomicAdd(&deg[(int)ei[(size_t)E + e]], 1.0f);
}

__global__ __launch_bounds__(256) void k_dinv(float* __restrict__ d, int N) {
    int i = blockIdx.x * blockDim.x + threadIdx.x;
    if (i < N) { float v = d[i]; d[i] = v > 0.0f ? rsqrtf(v) : 0.0f; }
}

// ---------------------------------------------------------------- GEMM: H = X * W
// Grid: N/16 blocks of 256 threads (8 waves). Wave w computes the 16x16 tile
// (blockIdx.x*16, w*16). K=128 -> 4 x v_wmma_f32_16x16x32_bf16, f32 accumulate.
// A fragment (16x32 bf16): lane l (row = l&15, koff = (l>>4)*8) holds
//   elems 0..7  = A[row][kb+koff .. +7],  elems 8..15 = A[row][kb+koff+16 .. +23]
// B fragment mirrors with col = l&15. D: lane l, vgpr r -> (8*(l>>4)+r, l&15).
__global__ __launch_bounds__(256) void k_gemm(const float* __restrict__ X,
                                              const float* __restrict__ W,
                                              float* __restrict__ H) {
    const int lane = threadIdx.x & 31;
    const int wave = threadIdx.x >> 5;
    const int row0 = blockIdx.x * 16;
    const int col0 = wave * 16;
    const int lrow = lane & 15;
    const int koff = (lane >> 4) * 8;
    const int col  = col0 + lrow;

    v8f acc = {};
    const float* xrow = X + (size_t)(row0 + lrow) * DIM;
#pragma unroll
    for (int kb = 0; kb < DIM; kb += 32) {
        v16bf a, b;
        const float* xa = xrow + kb + koff;
#pragma unroll
        for (int i = 0; i < 8; ++i) {
            a[i]     = (__bf16)xa[i];
            a[i + 8] = (__bf16)xa[i + 16];
        }
        const float* wb = W + (size_t)(kb + koff) * DIM + col;
#pragma unroll
        for (int i = 0; i < 8; ++i) {
            b[i]     = (__bf16)wb[(size_t)i * DIM];
            b[i + 8] = (__bf16)wb[(size_t)(i + 16) * DIM];
        }
        acc = __builtin_amdgcn_wmma_f32_16x16x32_bf16(
            /*neg_a=*/false, a, /*neg_b=*/false, b,
            /*c_mod=*/(short)0, acc, /*reuse_a=*/false, /*reuse_b=*/false);
    }
    const int mhi = (lane >> 4) * 8;
#pragma unroll
    for (int r = 0; r < 8; ++r)
        H[(size_t)(row0 + mhi + r) * DIM + col] = acc[r];
}

// ---------------------------------------------------------------- edge scatter
// One wave32 per (edge or self-loop). Lane reads float4 of H[src] (coalesced
// 512B per wave, L2-resident) and does 4 global_atomic_add_f32 into out[dst].
__global__ __launch_bounds__(256) void k_aggregate(const float* __restrict__ H,
                                                   const long long* __restrict__ ei,
                                                   const float* __restrict__ dinv,
                                                   float* __restrict__ out,
                                                   int E, int N) {
    const int wid  = (int)(((size_t)blockIdx.x * blockDim.x + threadIdx.x) >> 5);
    const int lane = threadIdx.x & 31;
    if (wid >= E + N) return;

    int s, d; float nrm;
    if (wid < E) {
        s = (int)ei[wid];
        d = (int)ei[(size_t)E + wid];
        nrm = dinv[s] * dinv[d];
    } else {                       // self-loop
        s = d = wid - E;
        float v = dinv[s];
        nrm = v * v;
    }
    const float4 hv = *(const float4*)(H + (size_t)s * DIM + lane * 4);
    float* o = out + (size_t)d * DIM + lane * 4;
    atomicAdd(o + 0, nrm * hv.x);
    atomicAdd(o + 1, nrm * hv.y);
    atomicAdd(o + 2, nrm * hv.z);
    atomicAdd(o + 3, nrm * hv.w);
}

// ---------------------------------------------------------------- bias + LeakyReLU
__global__ __launch_bounds__(256) void k_bias_lrelu(const float* __restrict__ in,
                                                    const float* __restrict__ bias,
                                                    float* __restrict__ out, size_t n) {
    size_t i = (size_t)blockIdx.x * blockDim.x + threadIdx.x;
    if (i < n) {
        float v = in[i] + bias[i & (DIM - 1)];
        out[i] = v >= 0.0f ? v : NEG_SLOPE * v;
    }
}

// ---------------------------------------------------------------- driver
extern "C" void kernel_launch(void* const* d_in, const int* in_sizes, int n_in,
                              void* d_out, int out_size, void* d_ws, size_t ws_size,
                              hipStream_t stream) {
    const float*     x  = (const float*)d_in[0];
    const long long* ei = (const long long*)d_in[1];
    const float*     W1 = (const float*)d_in[2];
    const float*     b1 = (const float*)d_in[3];
    const float*     W2 = (const float*)d_in[4];
    const float*     b2 = (const float*)d_in[5];
    const float*     W3 = (const float*)d_in[6];
    const float*     b3 = (const float*)d_in[7];

    const int N = in_sizes[0] / DIM;
    const int E = in_sizes[1] / 2;
    float* out = (float*)d_out;          // also reused as the scatter accumulator

    // workspace layout: dinv[N] | h[N*DIM] | xin[N*DIM]
    float* wsf  = (float*)d_ws;
    float* dinv = wsf;
    size_t pad  = ((size_t)N + 63) & ~(size_t)63;
    float* h    = wsf + pad;
    float* xin  = h + (size_t)N * DIM;

    const size_t   nf      = (size_t)N * DIM;
    const unsigned mTiles  = (unsigned)(N / 16);          // 50000/16 = 3125 exact
    const unsigned nBlk    = (unsigned)(((size_t)N + 255) / 256);
    const unsigned eBlk    = (unsigned)(((size_t)E + 255) / 256);
    const unsigned fBlk    = (unsigned)((nf + 255) / 256);
    const unsigned aggBlk  = (unsigned)(((size_t)(E + N) * 32 + 255) / 256);
    dim3 blk(256);

    // symmetric normalization: deg = 1 (self-loop) + in-degree; dinv = deg^-1/2
    k_fill<<<nBlk, blk, 0, stream>>>(dinv, (size_t)N, 1.0f);
    k_deg <<<eBlk, blk, 0, stream>>>(ei, dinv, E);
    k_dinv<<<nBlk, blk, 0, stream>>>(dinv, N);

    // layer 1: x -> xin
    k_gemm      <<<mTiles, blk, 0, stream>>>(x, W1, h);
    k_fill      <<<fBlk,   blk, 0, stream>>>(out, nf, 0.0f);
    k_aggregate <<<aggBlk, blk, 0, stream>>>(h, ei, dinv, out, E, N);
    k_bias_lrelu<<<fBlk,   blk, 0, stream>>>(out, b1, xin, nf);

    // layer 2: xin -> xin
    k_gemm      <<<mTiles, blk, 0, stream>>>(xin, W2, h);
    k_fill      <<<fBlk,   blk, 0, stream>>>(out, nf, 0.0f);
    k_aggregate <<<aggBlk, blk, 0, stream>>>(h, ei, dinv, out, E, N);
    k_bias_lrelu<<<fBlk,   blk, 0, stream>>>(out, b2, xin, nf);

    // layer 3: xin -> out (in-place bias+lrelu)
    k_gemm      <<<mTiles, blk, 0, stream>>>(xin, W3, h);
    k_fill      <<<fBlk,   blk, 0, stream>>>(out, nf, 0.0f);
    k_aggregate <<<aggBlk, blk, 0, stream>>>(h, ei, dinv, out, E, N);
    k_bias_lrelu<<<fBlk,   blk, 0, stream>>>(out, b3, out, nf);
}
